// BilateralSlice_17686675325316
// MI455X (gfx1250) — matrix operand order repaired and checked by reading, not compiled, not run on hardware
//
#include <hip/hip_runtime.h>
#include <hip/hip_bf16.h>

// ---------------------------------------------------------------------------
// Bilateral slice + affine apply (HDRNet), WMMA-accelerated for gfx1250.
//
// grid : [B=4, GH=16, GW=16, GD=8, GC=12] f32
// guide: [B, H=1024, W=1024]              f32
// inp  : [B, H, W, CIN=3]                 f32
// out  : [B, H, W, NOUT=3]                f32
//
// Key structural fact: one grid cell spans 64x64 pixels, so every 32x32
// aligned pixel quadrant shares a single (fx, fy) 2x2 xy-neighborhood; only
// the z (guide) coordinate varies per pixel. The trilinear gather is then an
// exact 16x16x32 matmul:  coeffs[16px x 12ch] = W[16 x 32] * G[32 x 12],
// K = (dy,dx,z) in 2x2x8 -> V_WMMA_F32_16X16X32_F16.
// ---------------------------------------------------------------------------

typedef __attribute__((ext_vector_type(16))) _Float16 v16h;
typedef __attribute__((ext_vector_type(8)))  float    v8f;

#define BATCH 4
#define IMH   1024
#define IMW   1024
#define GHH   16
#define GWW   16
#define GDD   8
#define GCC   12
#define CELL  64.0f   // pixels per grid cell (1024/16)

__global__ __launch_bounds__(256)
void bilateral_slice_wmma(const float* __restrict__ grid,
                          const float* __restrict__ guide,
                          const float* __restrict__ inp,
                          float* __restrict__ out)
{
    // Per-wave coefficient transpose tile: 32 pixels x 16 channels f32.
    __shared__ float smem[8][32][16];   // 16 KB

    const int tid  = threadIdx.x;
    const int wave = tid >> 5;
    const int lane = tid & 31;
    const int lo16 = lane & 15;
    const int hi   = lane >> 4;         // A: dx selector / B: dx selector / C: M-half

    // block -> (batch, quadrant-y, quadrant-x); 32x32-pixel quadrants
    int blk = blockIdx.x;
    const int qx = blk & 31; blk >>= 5;
    const int qy = blk & 31; blk >>= 5;
    const int b  = blk;

    const int X0 = qx << 5;
    const int Y0 = qy << 5;

    // Quadrant-constant cell coordinates (constant across all 32x32 pixels).
    const int fx = (int)floorf((X0 + 0.5f) * (1.0f / CELL) - 0.5f);
    const int fy = (int)floorf((Y0 + 0.5f) * (1.0f / CELL) - 0.5f);
    const int xi0 = max(fx, 0),     xi1 = min(fx + 1, GWW - 1);
    const int yi0 = max(fy, 0),     yi1 = min(fy + 1, GHH - 1);

    // ------------------------------------------------------------------
    // B matrix (grid slab), shared by the whole quadrant, loaded once.
    // Lane holds channel column c = lane%16; K-half kh = lane>=16 (dx).
    // Element e: K = kh*16 + e -> dx=kh, dy=(e>=8), z=e%8.
    // Grid is 384 KB -> permanently L2-resident; these are cheap hits.
    // ------------------------------------------------------------------
    v16h bmat = {};
    {
        const int xi = hi ? xi1 : xi0;
        const float* g0 = grid + (((size_t)(b * GHH + yi0) * GWW + xi) * GDD) * GCC + lo16;
        const float* g1 = grid + (((size_t)(b * GHH + yi1) * GWW + xi) * GDD) * GCC + lo16;
        if (lo16 < GCC) {
            #pragma unroll
            for (int z = 0; z < 8; ++z) {
                bmat[z]     = (_Float16)g0[z * GCC];   // dy = 0
                bmat[z + 8] = (_Float16)g1[z * GCC];   // dy = 1
            }
        }
    }

    // Each wave owns 4 image rows of this quadrant.
    for (int it = 0; it < 4; ++it) {
        const int y  = Y0 + (wave << 2) + it;
        const float gy  = (y + 0.5f) * (1.0f / CELL);
        const float yw0 = fmaxf(1.0f - fabsf((float)fy + 0.5f - gy), 0.0f);
        const float yw1 = fmaxf(1.0f - fabsf((float)fy + 1.5f - gy), 0.0f);

        // Two 16-pixel groups cover the 32-pixel row segment.
        #pragma unroll
        for (int g = 0; g < 2; ++g) {
            // A matrix: lane (and lane+16) own pixel m = lane%16 of this group;
            // lane's K-subset corresponds to dx = hi. v16h elems [0..7] = dy0,
            // elems [8..15] = dy1, z = e%8 (K = 16*hi + e, same order as B).
            const int   m   = lo16;
            const int   x   = X0 + (g << 4) + m;
            const float gxv = (x + 0.5f) * (1.0f / CELL);
            const float xw  = fmaxf(1.0f - fabsf((float)(fx + hi) + 0.5f - gxv), 0.0f);

            const float gv = guide[((size_t)b * IMH + y) * IMW + x];
            const float gz = fminf(fmaxf(gv, 0.0f), 1.0f) * (float)GDD;
            const float fz = floorf(gz - 0.5f);
            const float w0 = fmaxf(1.0f - fabsf(fz + 0.5f - gz), 0.0f);
            const float w1 = fmaxf(1.0f - fabsf(fz + 1.5f - gz), 0.0f);
            const int  zi0 = min(max((int)fz,     0), GDD - 1);
            const int  zi1 = min(max((int)fz + 1, 0), GDD - 1);

            const float s0 = xw * yw0;
            const float s1 = xw * yw1;

            v16h amat;
            #pragma unroll
            for (int z = 0; z < 8; ++z) {
                const float zz = ((z == zi0) ? w0 : 0.0f) + ((z == zi1) ? w1 : 0.0f);
                amat[z]     = (_Float16)(s0 * zz);
                amat[z + 8] = (_Float16)(s1 * zz);
            }

            // coeffs[16px x 16ch] = W * G  (single K=32 step, C starts at 0)
            v8f cacc = {};
            cacc = __builtin_amdgcn_wmma_f32_16x16x32_f16(
                /*neg_a=*/false, amat, /*neg_b=*/false, bmat,
                /*c_mod=*/(short)0, cacc, /*reuse_a=*/false, /*reuse_b=*/false);

            // C layout: lane holds channel N=lane%16, rows M = r + 8*hi.
            #pragma unroll
            for (int r = 0; r < 8; ++r) {
                smem[wave][(g << 4) + r + (hi << 3)][lo16] = cacc[r];
            }
        }

        // Within-wave LDS transpose: DS ops are in-order per wave; make the
        // dependency explicit and stop compiler reordering.
        asm volatile("s_wait_dscnt 0" ::: "memory");

        // Affine apply: all 32 lanes, one pixel each (coalesced 12B IO).
        {
            const int    p   = lane;
            const int    x   = X0 + p;
            const float* cf  = &smem[wave][p][0];
            const float4 c0  = *(const float4*)(cf + 0);
            const float4 c1  = *(const float4*)(cf + 4);
            const float4 c2  = *(const float4*)(cf + 8);

            const size_t pix = ((size_t)b * IMH + y) * IMW + x;
            const float* ip  = inp + pix * 3;
            const float  r   = ip[0];
            const float  gch = ip[1];
            const float  bl  = ip[2];

            float* op = out + pix * 3;
            op[0] = c0.x * r + c0.y * gch + c0.z * bl + c0.w;
            op[1] = c1.x * r + c1.y * gch + c1.z * bl + c1.w;
            op[2] = c2.x * r + c2.y * gch + c2.z * bl + c2.w;
        }
    }
}

extern "C" void kernel_launch(void* const* d_in, const int* in_sizes, int n_in,
                              void* d_out, int out_size, void* d_ws, size_t ws_size,
                              hipStream_t stream) {
    (void)in_sizes; (void)n_in; (void)d_ws; (void)ws_size; (void)out_size;
    const float* grid  = (const float*)d_in[0];
    const float* guide = (const float*)d_in[1];
    const float* inp   = (const float*)d_in[2];
    float*       out   = (float*)d_out;

    // blocks = B * (H/32) * (W/32) = 4 * 32 * 32 = 4096, 256 threads (8 waves)
    dim3 gridDim(BATCH * (IMH / 32) * (IMW / 32));
    dim3 blockDim(256);
    bilateral_slice_wmma<<<gridDim, blockDim, 0, stream>>>(grid, guide, inp, out);
}
